// NGP_62732292325674
// MI455X (gfx1250) — matrix operand order repaired and checked by reading, not compiled, not run on hardware
//
#include <hip/hip_runtime.h>
#include <hip/hip_bf16.h>
#include <cmath>

typedef __attribute__((ext_vector_type(16))) _Float16     v16h;
typedef __attribute__((ext_vector_type(8)))  float        v8f;
typedef __attribute__((ext_vector_type(4)))  unsigned int u32x4;

#define NLVL 16
#define HSIZE (1u << 19)
#define HMASK (HSIZE - 1u)
#define PR1 2654435761u
#define PR2 805459861u
#define PR3 3674653429u
#define BLOCK 128
#define FSTR 72   // LDS row stride in halves (>=64, keeps 16B alignment: 144B rows)

struct ResArr { float r[NLVL]; };

union FragH {
  v16h     v;
  _Float16 h[16];
  u32x4    u4[2];
};

// --- Pre-swizzle W1 (32x64) and W2 (64x64) f32 -> f16 WMMA B-fragments in ws.
// B 32x16 layout: lane holds column (16*ct + lane&15), 16 consecutive K values
// starting at klo = (lane<16 ? 0 : 16) (+32*kt for W2's second K-tile).
__global__ void ngp_prep_weights(const float* __restrict__ W1,
                                 const float* __restrict__ W2,
                                 _Float16* __restrict__ wf) {
  const int t    = threadIdx.x;      // 256 threads
  const int lane = t & 31;
  const int colL = lane & 15;
  const int kloL = (lane < 16) ? 0 : 16;
  if (t < 128) {                      // W1: 4 fragments (ct = 0..3)
    const int ct   = t >> 5;
    const int coln = ct * 16 + colL;
    _Float16* dst  = wf + (ct * 32 + lane) * 16;
#pragma unroll
    for (int j = 0; j < 16; ++j)
      dst[j] = (_Float16)W1[(kloL + j) * 64 + coln];
  }
  {                                   // W2: 8 fragments (fi = kt*4 + ct)
    const int fi   = t >> 5;
    const int kt   = fi >> 2, ct = fi & 3;
    const int coln = ct * 16 + colL;
    const int klo  = kt * 32 + kloL;
    _Float16* dst  = wf + 2048 + (fi * 32 + lane) * 16;
#pragma unroll
    for (int j = 0; j < 16; ++j)
      dst[j] = (_Float16)W2[(klo + j) * 64 + coln];
  }
}

__global__ void __launch_bounds__(BLOCK)
ngp_fused(const float* __restrict__ inp, const float* __restrict__ latent,
          const float* __restrict__ tables,
          const float* __restrict__ b1, const float* __restrict__ b2,
          const float* __restrict__ W3, const float* __restrict__ b3,
          const _Float16* __restrict__ wf, float* __restrict__ out,
          int nPer, ResArr R) {
  __shared__ __align__(16) _Float16 sF[BLOCK * FSTR];  // features, then h2
  __shared__ __align__(16) _Float16 sH[BLOCK * FSTR];  // h1
  const int tid = threadIdx.x;
  const int p   = blockIdx.x * BLOCK + tid;

  // ---------- Phase 1: hash-grid encode (gather-bound, tables live in L2) ----
  {
    const float x0 = (inp[3 * p + 0] + 1.f) * 0.5f;
    const float x1 = (inp[3 * p + 1] + 1.f) * 0.5f;
    const float x2 = (inp[3 * p + 2] + 1.f) * 0.5f;
    const float x3 = (latent[p / nPer] + 1.f) * 0.5f;
    _Float16* fr = &sF[tid * FSTR];
    for (int l = 0; l < NLVL; ++l) {
      const float res = R.r[l];
      float q, fq;
      q = x0 * res; fq = floorf(q); const float w0 = q - fq; const unsigned a0 = (unsigned)fq;
      q = x1 * res; fq = floorf(q); const float w1 = q - fq; const unsigned a1 = (unsigned)fq;
      q = x2 * res; fq = floorf(q); const float w2 = q - fq; const unsigned a2 = (unsigned)fq;
      q = x3 * res; fq = floorf(q); const float w3 = q - fq; const unsigned a3 = (unsigned)fq;
      const unsigned hx0 = a0,       hx1 = a0 + 1u;
      const unsigned hy0 = a1 * PR1, hy1 = hy0 + PR1;
      const unsigned hz0 = a2 * PR2, hz1 = hz0 + PR2;
      const unsigned hw0 = a3 * PR3, hw1 = hw0 + PR3;
      const float u0 = 1.f - w0, u1 = 1.f - w1, u2 = 1.f - w2, u3 = 1.f - w3;
      const float* tab = tables + (size_t)l * (size_t)HSIZE * 2u;
      float f0 = 0.f, f1 = 0.f;
#pragma unroll
      for (int c = 0; c < 16; ++c) {
        unsigned idx = ((c & 1) ? hx1 : hx0) ^ ((c & 2) ? hy1 : hy0) ^
                       ((c & 4) ? hz1 : hz0) ^ ((c & 8) ? hw1 : hw0);
        idx &= HMASK;
        const float wt = ((c & 1) ? w0 : u0) * ((c & 2) ? w1 : u1) *
                         ((c & 4) ? w2 : u2) * ((c & 8) ? w3 : u3);
        const float2 tv = *reinterpret_cast<const float2*>(tab + 2u * idx);
        f0 = fmaf(wt, tv.x, f0);
        f1 = fmaf(wt, tv.y, f1);
      }
      fr[2 * l]     = (_Float16)f0;
      fr[2 * l + 1] = (_Float16)f1;
    }
  }
  __syncthreads();

  const int wv   = tid >> 5;
  const int lane = tid & 31;
  const int colL = lane & 15;
  const int hs   = (lane < 16) ? 0 : 1;  // lane half-selector
  const int kb   = hs * 16;              // A-fragment K base within a 32-chunk

  // ---------- Phase 2: layer 1  (128x32) @ (32x64), K=32 = one WMMA ----------
#pragma unroll
  for (int mt = 0; mt < 2; ++mt) {
    const int m0 = (wv * 2 + mt) * 16;
    const _Float16* rp = &sF[(m0 + colL) * FSTR];
    FragH a;
    a.u4[0] = *reinterpret_cast<const u32x4*>(rp + kb);        // K kb..kb+7
    a.u4[1] = *reinterpret_cast<const u32x4*>(rp + kb + 8);    // K kb+8..kb+15
#pragma unroll
    for (int ct = 0; ct < 4; ++ct) {
      FragH b;
      b.v = *reinterpret_cast<const v16h*>(wf + (ct * 32 + lane) * 16);
      const float bias = b1[ct * 16 + colL];
      v8f c;
#pragma unroll
      for (int r = 0; r < 8; ++r) c[r] = bias;
      c = __builtin_amdgcn_wmma_f32_16x16x32_f16(false, a.v, false, b.v,
                                                 (short)0, c, false, false);
#pragma unroll
      for (int r = 0; r < 8; ++r) {
        const float v = fmaxf(c[r], 0.f);
        sH[(m0 + r + hs * 8) * FSTR + ct * 16 + colL] = (_Float16)v;
      }
    }
  }
  __syncthreads();

  // ---------- Phase 3: layer 2  (128x64) @ (64x64), K=64 = two WMMAs ---------
#pragma unroll
  for (int mt = 0; mt < 2; ++mt) {
    const int m0 = (wv * 2 + mt) * 16;
    const _Float16* rp = &sH[(m0 + colL) * FSTR];
    FragH a0, a1;
    a0.u4[0] = *reinterpret_cast<const u32x4*>(rp + kb);
    a0.u4[1] = *reinterpret_cast<const u32x4*>(rp + kb + 8);
    a1.u4[0] = *reinterpret_cast<const u32x4*>(rp + 32 + kb);
    a1.u4[1] = *reinterpret_cast<const u32x4*>(rp + 32 + kb + 8);
#pragma unroll
    for (int ct = 0; ct < 4; ++ct) {
      FragH bA, bB;
      bA.v = *reinterpret_cast<const v16h*>(wf + 2048 + ((0 * 4 + ct) * 32 + lane) * 16);
      bB.v = *reinterpret_cast<const v16h*>(wf + 2048 + ((1 * 4 + ct) * 32 + lane) * 16);
      const float bias = b2[ct * 16 + colL];
      v8f c;
#pragma unroll
      for (int r = 0; r < 8; ++r) c[r] = bias;
      c = __builtin_amdgcn_wmma_f32_16x16x32_f16(false, a0.v, false, bA.v,
                                                 (short)0, c, false, false);
      c = __builtin_amdgcn_wmma_f32_16x16x32_f16(false, a1.v, false, bB.v,
                                                 (short)0, c, false, false);
#pragma unroll
      for (int r = 0; r < 8; ++r) {
        const float v = fmaxf(c[r], 0.f);
        sF[(m0 + r + hs * 8) * FSTR + ct * 16 + colL] = (_Float16)v;
      }
    }
  }
  __syncthreads();

  // ---------- Phase 4: layer 3 (64 -> 1), per-thread dot product -------------
  {
    float acc = b3[0];
    const _Float16* hp = &sF[tid * FSTR];
#pragma unroll
    for (int k = 0; k < 64; ++k) acc = fmaf((float)hp[k], W3[k], acc);
    out[p] = acc;
  }
}

extern "C" void kernel_launch(void* const* d_in, const int* in_sizes, int n_in,
                              void* d_out, int out_size, void* d_ws, size_t ws_size,
                              hipStream_t stream) {
  const float* inp    = (const float*)d_in[0];  // (B, N, 3)
  const float* latent = (const float*)d_in[1];  // (B, 1)
  const float* tables = (const float*)d_in[2];  // (16, 2^19, 2)
  const float* W1     = (const float*)d_in[3];  // (32, 64)
  const float* b1     = (const float*)d_in[4];  // (64,)
  const float* W2     = (const float*)d_in[5];  // (64, 64)
  const float* b2     = (const float*)d_in[6];  // (64,)
  const float* W3     = (const float*)d_in[7];  // (64, 1)
  const float* b3     = (const float*)d_in[8];  // (1,)
  float*    out = (float*)d_out;
  _Float16* wf  = (_Float16*)d_ws;              // 6144 halves = 12 KB used

  const int B    = in_sizes[1];                 // latent has B elements
  const int P    = out_size;                    // B * N points
  const int nPer = P / B;                       // N per batch

  ResArr R;
  double s = 1.0;
  for (int l = 0; l < NLVL; ++l) { R.r[l] = (float)std::floor(16.0 * s); s *= 1.3819; }

  ngp_prep_weights<<<1, 256, 0, stream>>>(W1, W2, wf);
  ngp_fused<<<P / BLOCK, BLOCK, 0, stream>>>(inp, latent, tables, b1, b2, W3, b3,
                                             wf, out, nPer, R);
}
